// KpiEncoder_14663018348754
// MI455X (gfx1250) — matrix operand order repaired and checked by reading, not compiled, not run on hardware
//
#include <hip/hip_runtime.h>

// ---------------------------------------------------------------------------
// KPI encoder for MI455X (gfx1250, wave32, WMMA).
// Pipeline: conv1(relu) -> conv2(relu, transpose) -> 2x [EdgeConv + FFN]
// EdgeConv linearized: h[n,k] = U[n] + V[idx[n,k]] - V[n] + b  (U=X W1^T, V=X W2^T)
// All dense GEMMs + kNN Gram matrix use v_wmma_f32_16x16x32_f16, with
// double-buffered operand tiles to overlap global loads with WMMA issue.
// ---------------------------------------------------------------------------

typedef __attribute__((ext_vector_type(16))) _Float16 v16h;
typedef __attribute__((ext_vector_type(8)))  float    v8f;

#define BATCH 16
#define WLEN  2048
#define NPTS  2048
#define CIN   100
#define C1    64
#define C2    128
#define KNN   9
#define NROWS (BATCH * NPTS)   // 32768 flattened (b,n) rows

__device__ __forceinline__ float gelu_f(float x) {
  return 0.5f * x * (1.0f + erff(x * 0.7071067811865476f));
}

// ---- WMMA f16 operand loaders (16x16x32). Activations/weights stored
// row-major [row][K=128] so every operand fetch is contiguous 16B/32B. ----

// A tile (16xK): lane holds row M=l&15. halves 0..7 = K 32q+8g.. ; 8..15 = K 32q+16+8g..
__device__ __forceinline__ v16h load_a(const _Float16* row, int g, int q) {
  v16h a;
  uint4* u = reinterpret_cast<uint4*>(&a);
  u[0] = *reinterpret_cast<const uint4*>(row + 32 * q + 8 * g);
  u[1] = *reinterpret_cast<const uint4*>(row + 32 * q + 16 + 8 * g);
  return a;
}
// B tile (Kx16) from [N][K]-major memory: lane holds col N=l&15. halves 0..7 = K 32q+8g.. ; 8..15 = K 32q+16+8g..
__device__ __forceinline__ v16h load_b(const _Float16* row, int g, int q) {
  v16h b;
  uint4* u = reinterpret_cast<uint4*>(&b);
  u[0] = *reinterpret_cast<const uint4*>(row + 32 * q + 8 * g);
  u[1] = *reinterpret_cast<const uint4*>(row + 32 * q + 16 + 8 * g);
  return b;
}

__device__ __forceinline__ v8f wmma_f16(v16h a, v16h b, v8f c) {
  return __builtin_amdgcn_wmma_f32_16x16x32_f16(false, a, false, b, (short)0, c,
                                                false, false);
}

// sorted top-9 insertion (value desc, ties -> lower index, matching lax.top_k)
__device__ __forceinline__ void top9_update(const v8f& c, int mt, int g,
                                            float (&tv)[KNN], int (&ti)[KNN]) {
#pragma unroll
  for (int r = 0; r < 8; ++r) {
    float v = c[r];
    int ii = mt * 16 + 8 * g + r;
    if (v > tv[KNN - 1]) {
      tv[KNN - 1] = v;
      ti[KNN - 1] = ii;
#pragma unroll
      for (int j = KNN - 1; j >= 1; --j) {
        if (tv[j] > tv[j - 1]) {
          float tf = tv[j]; tv[j] = tv[j - 1]; tv[j - 1] = tf;
          int t2 = ti[j]; ti[j] = ti[j - 1]; ti[j - 1] = t2;
        }
      }
    }
  }
}

// ---------------------------------------------------------------------------
// conv1: (B,100,2048) -> relu -> (B,64,2048), causal k=3 d=1 (taps w-2,w-1,w)
// ---------------------------------------------------------------------------
__global__ void conv1_kernel(const float* __restrict__ ts,
                             const float* __restrict__ w,
                             const float* __restrict__ bias,
                             float* __restrict__ out) {
  int id = blockIdx.x * blockDim.x + threadIdx.x;  // B*C1*WLEN
  int wpos = id & (WLEN - 1);
  int rem = id >> 11;
  int o = rem & (C1 - 1);
  int b = rem >> 6;
  float acc = bias[o];
  const float* xb = ts + ((size_t)b * CIN) * WLEN + wpos;
  const float* wb = w + (size_t)o * CIN * 3;
  for (int c = 0; c < CIN; ++c) {
    const float* xp = xb + (size_t)c * WLEN;
    const float* wp = wb + c * 3;
    if (wpos >= 2) acc += wp[0] * xp[-2];
    if (wpos >= 1) acc += wp[1] * xp[-1];
    acc += wp[2] * xp[0];
  }
  out[id] = fmaxf(acc, 0.0f);
}

// ---------------------------------------------------------------------------
// conv2: (B,64,2048) -> relu -> transposed xt (B,2048,128), causal k=3 d=2
// ---------------------------------------------------------------------------
__global__ void conv2_kernel(const float* __restrict__ h1,
                             const float* __restrict__ w,
                             const float* __restrict__ bias,
                             float* __restrict__ xt) {
  int id = blockIdx.x * blockDim.x + threadIdx.x;  // B*C2*WLEN
  int wpos = id & (WLEN - 1);
  int rem = id >> 11;
  int o = rem & (C2 - 1);
  int b = rem >> 7;
  float acc = bias[o];
  const float* xb = h1 + ((size_t)b * C1) * WLEN + wpos;
  const float* wb = w + (size_t)o * C1 * 3;
  for (int c = 0; c < C1; ++c) {
    const float* xp = xb + (size_t)c * WLEN;
    const float* wp = wb + c * 3;
    if (wpos >= 4) acc += wp[0] * xp[-4];
    if (wpos >= 2) acc += wp[1] * xp[-2];
    acc += wp[2] * xp[0];
  }
  xt[((size_t)b * NPTS + wpos) * C2 + o] = fmaxf(acc, 0.0f);
}

// ---------------------------------------------------------------------------
// prep: per row compute L2 norm; emit f16 normalized rows (for kNN) + f16 rows
// ---------------------------------------------------------------------------
__global__ void prep_kernel(const float* __restrict__ xt,
                            _Float16* __restrict__ xnh,
                            _Float16* __restrict__ xh) {
  int r = blockIdx.x * blockDim.x + threadIdx.x;  // NROWS
  const float* x = xt + (size_t)r * C2;
  float s = 0.0f;
  for (int c = 0; c < C2; ++c) s += x[c] * x[c];
  float inv = 1.0f / fmaxf(sqrtf(s), 1e-12f);
  _Float16* pn = xnh + (size_t)r * C2;
  _Float16* ph = xh + (size_t)r * C2;
  for (int c = 0; c < C2; ++c) {
    float v = x[c];
    ph[c] = (_Float16)v;
    pn[c] = (_Float16)(v * inv);
  }
}

// ---------------------------------------------------------------------------
// kNN: per wave, 16 query points; stream 128 key tiles through WMMA Gram with
// double-buffered A tiles; per-lane sorted top-9; pairwise merge via LDS.
// ---------------------------------------------------------------------------
__global__ __launch_bounds__(32) void knn_kernel(const _Float16* __restrict__ xnh,
                                                 int* __restrict__ idx_out) {
  const int l = threadIdx.x;
  const int g = l >> 4;
  const int m15 = l & 15;
  const int b = blockIdx.x >> 7;         // 128 n-tiles per batch
  const int ntile = blockIdx.x & 127;
  const int n_base = ntile * 16;
  const _Float16* xb = xnh + (size_t)b * NPTS * C2;

  // B operand: query tile (held across the whole key loop)
  v16h bq[4];
  {
    const _Float16* qrow = xb + (size_t)(n_base + m15) * C2;
#pragma unroll
    for (int q = 0; q < 4; ++q) bq[q] = load_b(qrow, g, q);
  }

  float tv[KNN];
  int ti[KNN];
#pragma unroll
  for (int j = 0; j < KNN; ++j) { tv[j] = -3.0e38f; ti[j] = 0; }

  // double-buffered A tiles: loads for tile mt+1 overlap WMMA+top9 of tile mt
  v16h a0[4], a1[4];
  {
    const _Float16* krow = xb + (size_t)m15 * C2;  // tile 0
#pragma unroll
    for (int q = 0; q < 4; ++q) a0[q] = load_a(krow, g, q);
  }

  for (int mt = 0; mt < 128; mt += 2) {
    {  // prefetch tile mt+1 (always exists: 128 even)
      const _Float16* nrow = xb + (size_t)((mt + 1) * 16 + m15) * C2;
#pragma unroll
      for (int q = 0; q < 4; ++q) a1[q] = load_a(nrow, g, q);
    }
    v8f c = {};
#pragma unroll
    for (int q = 0; q < 4; ++q) c = wmma_f16(a0[q], bq[q], c);
    top9_update(c, mt, g, tv, ti);

    if (mt + 2 < 128) {  // prefetch tile mt+2
      const _Float16* nrow = xb + (size_t)((mt + 2) * 16 + m15) * C2;
#pragma unroll
      for (int q = 0; q < 4; ++q) a0[q] = load_a(nrow, g, q);
    }
    v8f c2 = {};
#pragma unroll
    for (int q = 0; q < 4; ++q) c2 = wmma_f16(a1[q], bq[q], c2);
    top9_update(c2, mt + 1, g, tv, ti);
  }

  __shared__ float sv[2][16][KNN];
  __shared__ int si[2][16][KNN];
#pragma unroll
  for (int j = 0; j < KNN; ++j) { sv[g][m15][j] = tv[j]; si[g][m15][j] = ti[j]; }
  __syncthreads();

  if (l < 16) {
    int pa = 0, pb = 0;
    int orow = (b * NPTS + n_base + l) * KNN;
    for (int j = 0; j < KNN; ++j) {
      float fa = sv[0][l][pa], fb = sv[1][l][pb];
      int ia = si[0][l][pa], ib = si[1][l][pb];
      bool ta = (fa > fb) || (fa == fb && ia < ib);
      int m = ta ? ia : ib;
      if (ta) ++pa; else ++pb;
      idx_out[orow + j] = b * NPTS + m;  // global row index for gathers
    }
  }
}

// ---------------------------------------------------------------------------
// GEMM: Y[NROWS x 128] = Xh[NROWS x 128] * Wh^T (+bias). 8 waves/block,
// each wave: 16 rows x 128 cols, K=128 => 32 v_wmma per wave. B (weight)
// tiles double-buffered across the 8 output tiles.
// Wh is row-major [out=128][K=128] (i.e. the natural 'oc' weight layout).
// ---------------------------------------------------------------------------
__global__ __launch_bounds__(256) void gemm128_kernel(
    const _Float16* __restrict__ X, const _Float16* __restrict__ Wh,
    const float* __restrict__ bias, float* __restrict__ Y) {
  const int l = threadIdx.x & 31;
  const int wv = threadIdx.x >> 5;
  const int g = l >> 4;
  const int m15 = l & 15;
  const int row_base = (blockIdx.x * 8 + wv) * 16;

  v16h A[4];
  {
    const _Float16* arow = X + (size_t)(row_base + m15) * C2;
#pragma unroll
    for (int q = 0; q < 4; ++q) A[q] = load_a(arow, g, q);
  }

  v16h b0[4], b1[4];
  {
    const _Float16* brow = Wh + (size_t)m15 * C2;  // tile t=0
#pragma unroll
    for (int q = 0; q < 4; ++q) b0[q] = load_b(brow, g, q);
  }

  for (int t = 0; t < 8; t += 2) {
    {  // prefetch weight tile t+1 (8 even, always exists)
      const _Float16* brow = Wh + (size_t)((t + 1) * 16 + m15) * C2;
#pragma unroll
      for (int q = 0; q < 4; ++q) b1[q] = load_b(brow, g, q);
    }
    {
      v8f c = {};
#pragma unroll
      for (int q = 0; q < 4; ++q) c = wmma_f16(A[q], b0[q], c);
      int col = t * 16 + m15;
      float bv = bias ? bias[col] : 0.0f;
      float* yp = Y + (size_t)(row_base + 8 * g) * C2 + col;
#pragma unroll
      for (int r = 0; r < 8; ++r) yp[(size_t)r * C2] = c[r] + bv;
    }
    if (t + 2 < 8) {  // prefetch weight tile t+2
      const _Float16* brow = Wh + (size_t)((t + 2) * 16 + m15) * C2;
#pragma unroll
      for (int q = 0; q < 4; ++q) b0[q] = load_b(brow, g, q);
    }
    {
      v8f c = {};
#pragma unroll
      for (int q = 0; q < 4; ++q) c = wmma_f16(A[q], b1[q], c);
      int col = (t + 1) * 16 + m15;
      float bv = bias ? bias[col] : 0.0f;
      float* yp = Y + (size_t)(row_base + 8 * g) * C2 + col;
#pragma unroll
      for (int r = 0; r < 8; ++r) yp[(size_t)r * C2] = c[r] + bv;
    }
  }
}

// ---------------------------------------------------------------------------
// EdgeConv combine: h_k = U[n]+V[j_k]-V[n]; affine, gelu, max_k; xt += max
// ---------------------------------------------------------------------------
__global__ void combine_kernel(float* __restrict__ xt,
                               const float* __restrict__ U,
                               const float* __restrict__ V,
                               const int* __restrict__ idx,
                               const float* __restrict__ gamma,
                               const float* __restrict__ beta,
                               _Float16* __restrict__ xh) {
  int id = blockIdx.x * blockDim.x + threadIdx.x;  // NROWS*128
  int o = id & (C2 - 1);
  int r = id >> 7;
  const int* ip = idx + (size_t)r * KNN;
  float u = U[id];
  float vs = V[id];
  float ga = gamma[o], be = beta[o];
  float best = -3.0e38f;
#pragma unroll
  for (int k = 0; k < KNN; ++k) {
    int j = ip[k];
    float h = u + V[(size_t)j * C2 + o] - vs;
    h = gelu_f(h * ga + be);
    best = fmaxf(best, h);
  }
  float out = xt[id] + best;
  xt[id] = out;
  xh[id] = (_Float16)out;
}

// gelu(T*gamma+beta) -> f16 for next GEMM
__global__ void act_kernel(const float* __restrict__ T,
                           const float* __restrict__ gamma,
                           const float* __restrict__ beta,
                           _Float16* __restrict__ Th) {
  int id = blockIdx.x * blockDim.x + threadIdx.x;
  int o = id & (C2 - 1);
  Th[id] = (_Float16)gelu_f(T[id] * gamma[o] + beta[o]);
}

// dst = T*gamma+beta  (f32)
__global__ void affine_kernel(const float* __restrict__ T,
                              const float* __restrict__ gamma,
                              const float* __restrict__ beta,
                              float* __restrict__ dst) {
  int id = blockIdx.x * blockDim.x + threadIdx.x;
  int o = id & (C2 - 1);
  dst[id] = T[id] * gamma[o] + beta[o];
}

// split gconv weight (128x256) -> W1h (o,c<128), W2h (o,c>=128), f16
__global__ void split_gconv_kernel(const float* __restrict__ gw,
                                   _Float16* __restrict__ w1h,
                                   _Float16* __restrict__ w2h) {
  int id = blockIdx.x * blockDim.x + threadIdx.x;  // 128*128
  int c = id & (C2 - 1);
  int o = id >> 7;
  w1h[id] = (_Float16)gw[(size_t)o * 256 + c];
  w2h[id] = (_Float16)gw[(size_t)o * 256 + 128 + c];
}

__global__ void cvt_f16_kernel(const float* __restrict__ src,
                               _Float16* __restrict__ dst, int n) {
  int id = blockIdx.x * blockDim.x + threadIdx.x;
  if (id < n) dst[id] = (_Float16)src[id];
}

// ---------------------------------------------------------------------------
// Host orchestration
// ---------------------------------------------------------------------------
extern "C" void kernel_launch(void* const* d_in, const int* in_sizes, int n_in,
                              void* d_out, int out_size, void* d_ws, size_t ws_size,
                              hipStream_t stream) {
  (void)in_sizes; (void)n_in; (void)out_size; (void)ws_size;
  const float* ts = (const float*)d_in[0];
  const float* conv1_w = (const float*)d_in[1];
  const float* conv1_b = (const float*)d_in[2];
  const float* conv2_w = (const float*)d_in[3];
  const float* conv2_b = (const float*)d_in[4];
  const float* gconv_w = (const float*)d_in[5];
  const float* gconv_b = (const float*)d_in[6];
  const float* gconv_gamma = (const float*)d_in[7];
  const float* gconv_beta = (const float*)d_in[8];
  const float* ffn1_w = (const float*)d_in[9];
  const float* ffn1_b = (const float*)d_in[10];
  const float* ffn1_gamma = (const float*)d_in[11];
  const float* ffn1_beta = (const float*)d_in[12];
  const float* ffn2_w = (const float*)d_in[13];
  const float* ffn2_b = (const float*)d_in[14];
  const float* ffn2_gamma = (const float*)d_in[15];
  const float* ffn2_beta = (const float*)d_in[16];

  char* ws = (char*)d_ws;
  // f32 buffers
  float* h1 = (float*)(ws + 0);                               //  8 MB
  float* xt = (float*)(ws + 8388608);                         // 16 MB
  float* U  = (float*)(ws + 25165824);                        // 16 MB (reused as ffn T)
  float* V  = (float*)(ws + 41943040);                        // 16 MB (reused as ffn T2)
  // f16 buffers
  _Float16* xh  = (_Float16*)(ws + 58720256);                 //  8 MB
  _Float16* xnh = (_Float16*)(ws + 67108864);                 //  8 MB
  _Float16* th  = (_Float16*)(ws + 75497472);                 //  8 MB
  int* idxb = (int*)(ws + 83886080);                          // ~1.2 MB
  _Float16* w1h = (_Float16*)(ws + 85065728);                 // 32 KB each
  _Float16* w2h = (_Float16*)(ws + 85098496);
  _Float16* f1h = (_Float16*)(ws + 85131264);
  _Float16* f2h = (_Float16*)(ws + 85164032);

  const int TPB = 256;
  const int NELEM = NROWS * C2;  // 4,194,304

  // conv stack
  conv1_kernel<<<(BATCH * C1 * WLEN) / TPB, TPB, 0, stream>>>(ts, conv1_w, conv1_b, h1);
  conv2_kernel<<<(BATCH * C2 * WLEN) / TPB, TPB, 0, stream>>>(h1, conv2_w, conv2_b, xt);

  for (int blk = 0; blk < 2; ++blk) {
    const float* gw = gconv_w + (size_t)blk * C2 * 256;
    const float* gb = gconv_b + blk * C2;
    const float* gga = gconv_gamma + blk * C2;
    const float* gbe = gconv_beta + blk * C2;

    // weight precision conversion (tiny)
    split_gconv_kernel<<<(C2 * C2) / TPB, TPB, 0, stream>>>(gw, w1h, w2h);
    cvt_f16_kernel<<<(C2 * C2) / TPB, TPB, 0, stream>>>(ffn1_w + (size_t)blk * C2 * C2, f1h, C2 * C2);
    cvt_f16_kernel<<<(C2 * C2) / TPB, TPB, 0, stream>>>(ffn2_w + (size_t)blk * C2 * C2, f2h, C2 * C2);

    // --- EdgeConv ---
    prep_kernel<<<NROWS / TPB, TPB, 0, stream>>>(xt, xnh, xh);
    knn_kernel<<<BATCH * (NPTS / 16), 32, 0, stream>>>(xnh, idxb);
    gemm128_kernel<<<NROWS / 128, TPB, 0, stream>>>(xh, w1h, gb, U);
    gemm128_kernel<<<NROWS / 128, TPB, 0, stream>>>(xh, w2h, nullptr, V);
    combine_kernel<<<NELEM / TPB, TPB, 0, stream>>>(xt, U, V, idxb, gga, gbe, xh);

    // --- FFN ---
    gemm128_kernel<<<NROWS / 128, TPB, 0, stream>>>(xh, f1h, ffn1_b + blk * C2, U);
    act_kernel<<<NELEM / TPB, TPB, 0, stream>>>(U, ffn1_gamma + blk * C2, ffn1_beta + blk * C2, th);
    gemm128_kernel<<<NROWS / 128, TPB, 0, stream>>>(th, f2h, ffn2_b + blk * C2, V);
    float* dst = (blk == 1) ? (float*)d_out : xt;
    affine_kernel<<<NELEM / TPB, TPB, 0, stream>>>(V, ffn2_gamma + blk * C2, ffn2_beta + blk * C2, dst);
  }
}
